// VQVAE_57681410785306
// MI455X (gfx1250) — matrix operand order repaired and checked by reading, not compiled, not run on hardware
//
#include <hip/hip_runtime.h>
#include <cstdint>
#include <cstddef>

// ---------------------------------------------------------------------------
// VQ-VAE encoder + vector quantizer for gfx1250 (MI455X).
// Activations bf16 NHWC.  Convs = implicit GEMM via v_wmma_f32_16x16x32_bf16.
// Each wave owns 16 pixels and ALL cout tiles (A loaded once per K-step,
// amortized over COUT/16 WMMAs).  A- and B-fragments software-pipelined.
// VQ: codebook staged in LDS (bank-padded), WMMA z@emb^T, shfl butterfly argmin.
// Workspace: ~321 MB.
// ---------------------------------------------------------------------------

typedef __attribute__((ext_vector_type(16))) __bf16 bfvec16;
typedef __attribute__((ext_vector_type(8)))  float  v8f;

union AFrag { uint4 q[2]; unsigned int w[8]; bfvec16 v; };
union BFrag { unsigned int w[8]; bfvec16 v; };

__device__ __forceinline__ unsigned short f2bf(float f) {
    unsigned int u = __builtin_bit_cast(unsigned int, f);
    unsigned int r = u + 0x7FFFu + ((u >> 16) & 1u);   // round-to-nearest-even
    return (unsigned short)(r >> 16);
}
__device__ __forceinline__ float bf2f(unsigned short h) {
    unsigned int u = ((unsigned int)h) << 16;
    return __builtin_bit_cast(float, u);
}
// ReLU on a packed pair of bf16 values (sign bit -> zero)
__device__ __forceinline__ unsigned int relu_pk(unsigned int x) {
    unsigned int lo = (x & 0x00008000u) ? 0u : (x & 0x0000FFFFu);
    unsigned int hi = (x & 0x80000000u) ? 0u : (x & 0xFFFF0000u);
    return lo | hi;
}

// ---------------------------------------------------------------------------
// Weight pack: OIHW f32 -> wp[kp][cout] dword = (bf16 K=2kp, bf16 K=2kp+1),
// K = tap*CIN + cin (tap-major) -- the conv kernel's K order.
// ---------------------------------------------------------------------------
__global__ void pack_w_k(const float* __restrict__ w, unsigned int* __restrict__ wp,
                         int CIN, int COUT, int KHKW, int total) {
    int t = blockIdx.x * blockDim.x + threadIdx.x;
    if (t >= total) return;
    int kp   = t / COUT;
    int cout = t - kp * COUT;
    int k0   = kp * 2;
    int tap  = k0 / CIN;
    int cin  = k0 - tap * CIN;           // even; cin+1 < CIN (CIN even)
    float a = w[((size_t)cout * CIN + cin)     * KHKW + tap];
    float b = w[((size_t)cout * CIN + cin + 1) * KHKW + tap];
    wp[(size_t)kp * COUT + cout] =
        (unsigned int)f2bf(a) | ((unsigned int)f2bf(b) << 16);
}

// ---------------------------------------------------------------------------
// conv1: 4x4 s2 p1, Cin=1 (K=16, too small for WMMA) -> direct conv.
// x: f32 (64,1,256,256); out: bf16 NHWC (64,128,128,64), ReLU.
// ---------------------------------------------------------------------------
__global__ __launch_bounds__(256) void conv1_k(const float* __restrict__ x,
                                               const float* __restrict__ w,
                                               const float* __restrict__ bias,
                                               unsigned short* __restrict__ out) {
    __shared__ float wl[1024];                    // (64,1,4,4)
    for (int i = threadIdx.x; i < 1024; i += 256) wl[i] = w[i];
    __syncthreads();

    const int p   = blockIdx.x * 256 + threadIdx.x;   // 1048576 exact
    const int b   = p >> 14;
    const int rem = p & 16383;
    const int oy  = rem >> 7;
    const int ox  = rem & 127;

    float acc[64];
    #pragma unroll
    for (int co = 0; co < 64; ++co) acc[co] = bias[co];

    for (int dy = 0; dy < 4; ++dy) {
        const int iy = oy * 2 - 1 + dy;
        if ((unsigned)iy >= 256u) continue;
        for (int dx = 0; dx < 4; ++dx) {
            const int ix = ox * 2 - 1 + dx;
            if ((unsigned)ix >= 256u) continue;
            const float xv = x[(size_t)(b * 256 + iy) * 256 + ix];
            const int tap = dy * 4 + dx;
            #pragma unroll
            for (int co = 0; co < 64; ++co)
                acc[co] = fmaf(xv, wl[co * 16 + tap], acc[co]);
        }
    }
    unsigned short* o = out + (size_t)p * 64;
    #pragma unroll
    for (int g = 0; g < 8; ++g) {
        union { unsigned short h[8]; uint4 q; } u;
        #pragma unroll
        for (int j = 0; j < 8; ++j) u.h[j] = f2bf(fmaxf(acc[g * 8 + j], 0.f));
        reinterpret_cast<uint4*>(o)[g] = u.q;
    }
}

// ---------------------------------------------------------------------------
// Implicit-GEMM conv, WMMA bf16.
//   M = B*OH*OW pixels, N = COUT, K = CIN*KH*KW (tap-major).
//   Block 256 thr = 8 waves; each wave: 16 pixels x ALL COUT/16 cout tiles.
//   A frag loaded once per K-step (2x b128/lane), reused for COUT/16 WMMAs.
//   A pipelined one K-step ahead; B pipelined one cout-tile ahead.
// ---------------------------------------------------------------------------
template<int CIN, int COUT, int KH, int KW, int S, int P,
         bool RIN, bool ROUT, bool RES>
__global__ __launch_bounds__(256) void conv_wmma_k(
    const unsigned short* __restrict__ in,
    const unsigned int*   __restrict__ wp,
    const float*          __restrict__ bias,
    unsigned short*       __restrict__ out,
    const unsigned short* __restrict__ res,
    int H, int W, int OH, int OW)
{
    constexpr int CPT = CIN / 32;       // K-chunks per tap
    constexpr int NIT = KH * KW * CPT;  // total K-steps
    constexpr int NT  = COUT / 16;      // cout tiles per wave

    const int tid    = threadIdx.x;
    const int wave   = tid >> 5;
    const int lane   = tid & 31;
    const int laneHi = lane >> 4;
    const int ln     = lane & 15;

    const int pixBase = blockIdx.x * 128 + wave * 16;
    const int myPix   = pixBase + ln;        // A-row pixel (both lane halves)
    const int ohw = OH * OW;
    const int b   = myPix / ohw;
    const int rp  = myPix - b * ohw;
    const int oy  = rp / OW;
    const int ox  = rp - oy * OW;
    const int iy0 = oy * S - P;
    const int ix0 = ox * S - P;

    // A fragment loader (contiguous K layout: lanes 0-15 K 0-15,
    // lanes 16-31 K 16-31, matching packed B).
    auto loadA = [&](int it, AFrag& a) {
        const int tap = it / CPT, c = it - tap * CPT;
        const int dy = tap / KW, dx = tap - (tap / KW) * KW;
        const int iy = iy0 + dy, ix = ix0 + dx;
        const bool valid = ((unsigned)iy < (unsigned)H) && ((unsigned)ix < (unsigned)W);
        if (valid) {
            const unsigned short* ab =
                in + ((size_t)(b * H + iy) * W + ix) * CIN + c * 32 + laneHi * 16;
            const uint4* ap = reinterpret_cast<const uint4*>(ab);
            a.q[0] = ap[0];          // K + {0..7}
            a.q[1] = ap[1];          // K + {8..15}
        } else {
            a.q[0] = make_uint4(0u, 0u, 0u, 0u);
            a.q[1] = make_uint4(0u, 0u, 0u, 0u);
        }
    };
    // B fragment loader from packed weights: VGPR j = dword (K=2kp,2kp+1), col ln.
    auto loadB = [&](BFrag& bf, const unsigned int* wrow, int nt) {
        #pragma unroll
        for (int j = 0; j < 8; ++j)
            bf.w[j] = wrow[(size_t)j * COUT + nt * 16];
    };

    v8f acc[NT];
    #pragma unroll
    for (int nt = 0; nt < NT; ++nt) acc[nt] = v8f{0.f,0.f,0.f,0.f,0.f,0.f,0.f,0.f};

    AFrag aN;
    loadA(0, aN);
    for (int it = 0; it < NIT; ++it) {
        AFrag aC = aN;
        if (it + 1 < NIT) loadA(it + 1, aN);         // prefetch next A
        if (RIN) {
            #pragma unroll
            for (int j = 0; j < 8; ++j) aC.w[j] = relu_pk(aC.w[j]);
        }
        // kpBase = it*16 + laneHi*8 (K order is tap-major, 16 kp rows per step)
        const unsigned int* wrow =
            wp + (size_t)(it * 16 + laneHi * 8) * COUT + ln;
        BFrag bC, bN;
        loadB(bC, wrow, 0);
        #pragma unroll
        for (int nt = 0; nt < NT; ++nt) {
            if (nt + 1 < NT) loadB(bN, wrow, nt + 1); // prefetch next B
            acc[nt] = __builtin_amdgcn_wmma_f32_16x16x32_bf16(
                false, aC.v, false, bC.v, (short)0, acc[nt], false, false);
            if (nt + 1 < NT) bC = bN;
        }
    }

    // Epilogue: C layout -> VGPR r: lanes 0-15 M=r, lanes 16-31 M=8+r.
    float bv[NT];
    #pragma unroll
    for (int nt = 0; nt < NT; ++nt) bv[nt] = bias[nt * 16 + ln];
    #pragma unroll
    for (int nt = 0; nt < NT; ++nt) {
        #pragma unroll
        for (int r = 0; r < 8; ++r) {
            const int m = r + laneHi * 8;
            const size_t oidx = (size_t)(pixBase + m) * COUT + nt * 16 + ln;
            float v = acc[nt][r] + bv[nt];
            if (ROUT) v = fmaxf(v, 0.f);
            if (RES)  v += bf2f(res[oidx]);
            out[oidx] = f2bf(v);
        }
    }
}

// ---------------------------------------------------------------------------
// VQ prep: codebook (512,64) f32 -> packed bf16 pairs embp[kp][code] + ||e||^2.
// ---------------------------------------------------------------------------
__global__ void vq_prep_k(const float* __restrict__ emb,
                          unsigned int* __restrict__ embp,
                          float* __restrict__ enorm) {
    const int code = blockIdx.x * blockDim.x + threadIdx.x;
    if (code >= 512) return;
    const float* e = emb + (size_t)code * 64;
    float s = 0.f;
    for (int d = 0; d < 64; ++d) s += e[d] * e[d];
    enorm[code] = s;
    for (int kp = 0; kp < 32; ++kp) {
        embp[(size_t)kp * 512 + code] =
            (unsigned int)f2bf(e[2 * kp]) | ((unsigned int)f2bf(e[2 * kp + 1]) << 16);
    }
}

// ---------------------------------------------------------------------------
// VQ argmin.  Codebook staged in LDS (row stride 516 dwords: 8-row lane-half
// offset lands 32 banks away -> conflict-free frag reads).  B frags ping-pong
// from LDS; dist = ||e||^2 - 2*dot; per-lane running min; 16-lane shfl_xor
// butterfly; lane 0 of each half writes indices for its 8 rows.
// Dynamic LDS: 32*516*4 + 512*4 = 68096 B.
// ---------------------------------------------------------------------------
#define VQ_LDS_STRIDE 516
#define VQ_LDS_BYTES  (32 * VQ_LDS_STRIDE * 4 + 512 * 4)

__global__ __launch_bounds__(256) void vq_argmin_k(
    const unsigned short* __restrict__ z,
    const unsigned int*   __restrict__ embp,
    const float*          __restrict__ enorm,
    int*                  __restrict__ idx)
{
    extern __shared__ char smem[];
    unsigned int* se = (unsigned int*)smem;                       // [32][516]
    float*        sn = (float*)(smem + 32 * VQ_LDS_STRIDE * 4);   // [512]

    const int tid = threadIdx.x;
    for (int i = tid; i < 32 * 512; i += 256) {
        const int kp = i >> 9, c = i & 511;
        se[kp * VQ_LDS_STRIDE + c] = embp[(size_t)kp * 512 + c];
    }
    for (int i = tid; i < 512; i += 256) sn[i] = enorm[i];
    __syncthreads();

    const int wave   = tid >> 5;
    const int lane   = tid & 31;
    const int laneHi = lane >> 4;
    const int ln     = lane & 15;
    const int pixBase = blockIdx.x * 128 + wave * 16;
    const int myPix   = pixBase + ln;

    AFrag a0, a1;
    const uint4* zp = reinterpret_cast<const uint4*>(
        z + (size_t)myPix * 64 + laneHi * 16);
    a0.q[0] = zp[0]; a0.q[1] = zp[1];   // K 0..31
    a1.q[0] = zp[4]; a1.q[1] = zp[5];   // K 32..63

    const int kb0 = laneHi * 8;         // chunk0 kp base
    const int kb1 = 16 + laneHi * 8;    // chunk1 kp base
    auto loadB = [&](BFrag& bf, int kb, int ct) {
        #pragma unroll
        for (int j = 0; j < 8; ++j)
            bf.w[j] = se[(kb + j) * VQ_LDS_STRIDE + ct * 16 + ln];
    };

    float minv[8]; int mini[8];
    #pragma unroll
    for (int r = 0; r < 8; ++r) { minv[r] = 3.4e38f; mini[r] = 0; }

    BFrag b0c, b1c, b0n, b1n;
    loadB(b0c, kb0, 0);
    loadB(b1c, kb1, 0);
    for (int ct = 0; ct < 32; ++ct) {
        if (ct + 1 < 32) { loadB(b0n, kb0, ct + 1); loadB(b1n, kb1, ct + 1); }
        v8f acc = {0.f, 0.f, 0.f, 0.f, 0.f, 0.f, 0.f, 0.f};
        acc = __builtin_amdgcn_wmma_f32_16x16x32_bf16(
            false, a0.v, false, b0c.v, (short)0, acc, false, false);
        acc = __builtin_amdgcn_wmma_f32_16x16x32_bf16(
            false, a1.v, false, b1c.v, (short)0, acc, false, false);

        const int n    = ct * 16 + ln;
        const float en = sn[n];
        #pragma unroll
        for (int r = 0; r < 8; ++r) {
            float cand = en - 2.f * acc[r];
            if (cand < minv[r]) { minv[r] = cand; mini[r] = n; }
        }
        if (ct + 1 < 32) { b0c = b0n; b1c = b1n; }
    }
    #pragma unroll
    for (int off = 8; off > 0; off >>= 1) {
        #pragma unroll
        for (int r = 0; r < 8; ++r) {
            float ov = __shfl_xor(minv[r], off, 32);
            int   oi = __shfl_xor(mini[r], off, 32);
            if (ov < minv[r] || (ov == minv[r] && oi < mini[r])) {
                minv[r] = ov; mini[r] = oi;
            }
        }
    }
    if (ln == 0) {
        #pragma unroll
        for (int r = 0; r < 8; ++r)
            idx[pixBase + r + laneHi * 8] = mini[r];
    }
}

// ---------------------------------------------------------------------------
// Gather: quantized[p][:] = emb[idx[p]][:] in f32 (float4 granularity).
// ---------------------------------------------------------------------------
__global__ void vq_gather_k(const int* __restrict__ idx,
                            const float* __restrict__ emb,
                            float4* __restrict__ out) {
    const int t = blockIdx.x * 256 + threadIdx.x;   // 262144*16 exact
    const int p = t >> 4, c = t & 15;
    out[t] = reinterpret_cast<const float4*>(emb)[idx[p] * 16 + c];
}

// ---------------------------------------------------------------------------
extern "C" void kernel_launch(void* const* d_in, const int* in_sizes, int n_in,
                              void* d_out, int out_size, void* d_ws, size_t ws_size,
                              hipStream_t stream) {
    (void)in_sizes; (void)n_in; (void)out_size; (void)ws_size;

    const float* x    = (const float*)d_in[0];
    const float* w1   = (const float*)d_in[1];
    const float* b1   = (const float*)d_in[2];
    const float* w2   = (const float*)d_in[3];
    const float* b2   = (const float*)d_in[4];
    const float* w3   = (const float*)d_in[5];
    const float* b3   = (const float*)d_in[6];
    const float* r1w1 = (const float*)d_in[7];
    const float* r1b1 = (const float*)d_in[8];
    const float* r1w2 = (const float*)d_in[9];
    const float* r1b2 = (const float*)d_in[10];
    const float* r2w1 = (const float*)d_in[11];
    const float* r2b1 = (const float*)d_in[12];
    const float* r2w2 = (const float*)d_in[13];
    const float* r2b2 = (const float*)d_in[14];
    const float* w4   = (const float*)d_in[15];
    const float* b4   = (const float*)d_in[16];
    const float* emb  = (const float*)d_in[17];

    char* ws = (char*)d_ws;
    size_t off = 0;
    auto take = [&](size_t bytes) -> char* {
        char* r = ws + off;
        off = (off + bytes + 255) & ~(size_t)255;
        return r;
    };
    unsigned short* act0 = (unsigned short*)take(134217728ull); // (64,128,128,64) bf16
    unsigned short* act1 = (unsigned short*)take(67108864ull);  // (64,64,64,128) bf16
    unsigned short* act2 = (unsigned short*)take(67108864ull);  // (64,64,64,128) bf16
    unsigned short* tbuf = (unsigned short*)take(16777216ull);  // (64,64,64,32)  bf16
    unsigned short* zbuf = (unsigned short*)take(33554432ull);  // (64,64,64,64)  bf16
    unsigned int* wp2  = (unsigned int*)take(512 * 128 * 4);
    unsigned int* wp3  = (unsigned int*)take(576 * 128 * 4);
    unsigned int* wpa1 = (unsigned int*)take(576 * 32 * 4);
    unsigned int* wpb1 = (unsigned int*)take(16 * 128 * 4);
    unsigned int* wpa2 = (unsigned int*)take(576 * 32 * 4);
    unsigned int* wpb2 = (unsigned int*)take(16 * 128 * 4);
    unsigned int* wp4  = (unsigned int*)take(64 * 64 * 4);
    unsigned int* ebp  = (unsigned int*)take(32 * 512 * 4);
    float* enorm = (float*)take(512 * 4);
    int*   idxb  = (int*)take(262144 * 4);

    // --- weight / codebook packing (tiny) ---
    pack_w_k<<<dim3((512 * 128 + 255) / 256), 256, 0, stream>>>(w2,   wp2,  64, 128, 16, 512 * 128);
    pack_w_k<<<dim3((576 * 128 + 255) / 256), 256, 0, stream>>>(w3,   wp3, 128, 128,  9, 576 * 128);
    pack_w_k<<<dim3((576 *  32 + 255) / 256), 256, 0, stream>>>(r1w1, wpa1,128,  32,  9, 576 * 32);
    pack_w_k<<<dim3(( 16 * 128 + 255) / 256), 256, 0, stream>>>(r1w2, wpb1, 32, 128,  1, 16 * 128);
    pack_w_k<<<dim3((576 *  32 + 255) / 256), 256, 0, stream>>>(r2w1, wpa2,128,  32,  9, 576 * 32);
    pack_w_k<<<dim3(( 16 * 128 + 255) / 256), 256, 0, stream>>>(r2w2, wpb2, 32, 128,  1, 16 * 128);
    pack_w_k<<<dim3(( 64 *  64 + 255) / 256), 256, 0, stream>>>(w4,   wp4, 128,  64,  1, 64 * 64);
    vq_prep_k<<<dim3(2), 256, 0, stream>>>(emb, ebp, enorm);

    // --- encoder (each block: 128 pixels x full COUT) ---
    conv1_k<<<dim3(4096), 256, 0, stream>>>(x, w1, b1, act0);
    conv_wmma_k<64, 128, 4, 4, 2, 1, false, true, false>
        <<<dim3(2048), 256, 0, stream>>>(act0, wp2, b2, act1, nullptr, 128, 128, 64, 64);
    conv_wmma_k<128, 128, 3, 3, 1, 1, false, false, false>
        <<<dim3(2048), 256, 0, stream>>>(act1, wp3, b3, act2, nullptr, 64, 64, 64, 64);
    conv_wmma_k<128, 32, 3, 3, 1, 1, true, true, false>
        <<<dim3(2048), 256, 0, stream>>>(act2, wpa1, r1b1, tbuf, nullptr, 64, 64, 64, 64);
    conv_wmma_k<32, 128, 1, 1, 1, 0, false, false, true>
        <<<dim3(2048), 256, 0, stream>>>(tbuf, wpb1, r1b2, act2, act2, 64, 64, 64, 64);
    conv_wmma_k<128, 32, 3, 3, 1, 1, true, true, false>
        <<<dim3(2048), 256, 0, stream>>>(act2, wpa2, r2b1, tbuf, nullptr, 64, 64, 64, 64);
    conv_wmma_k<32, 128, 1, 1, 1, 0, false, false, true>
        <<<dim3(2048), 256, 0, stream>>>(tbuf, wpb2, r2b2, act2, act2, 64, 64, 64, 64);
    conv_wmma_k<128, 64, 1, 1, 1, 0, true, false, false>
        <<<dim3(2048), 256, 0, stream>>>(act2, wp4, b4, zbuf, nullptr, 64, 64, 64, 64);

    // --- vector quantizer ---
    vq_argmin_k<<<dim3(2048), 256, VQ_LDS_BYTES, stream>>>(zbuf, ebp, enorm, idxb);
    vq_gather_k<<<dim3(16384), 256, 0, stream>>>(idxb, emb, (float4*)d_out);
}
